// SourceAttention_70317204570593
// MI455X (gfx1250) — matrix operand order repaired and checked by reading, not compiled, not run on hardware
//
#include <hip/hip_runtime.h>
#include <hip/hip_bf16.h>
#include <math.h>

// ---------------------------------------------------------------------------
// SourceAttention on gfx1250: all matmuls via v_wmma_f32_16x16x32_bf16.
// B=32, N=W*H=1024, L=512, C=D=768.
// Workspace (bf16): qp[B,1024,768] 48MB | kp[B,512,768] 24MB |
//                   vpT[B,768,512] 24MB | P[B,1024,512] 32MB |
//                   3x weight bf16 copies 3.4MB   (~131MB total; the whole
// intermediate set stays resident in MI455X's 192MB L2).
//
// v2: weights pre-converted to bf16 (kills half the v_cvt_pk_bf16_f32 in the
// projection hot loop) and every GEMM wave register-tiles 16x64 (1 A-fragment
// feeds 4 WMMAs) to amortize A-side loads/conversions 4x.
// ---------------------------------------------------------------------------

typedef __attribute__((ext_vector_type(16))) __bf16 v16bf;
typedef __attribute__((ext_vector_type(8)))  float  v8f;
typedef __attribute__((ext_vector_type(4)))  float  f32x4;

#define BATCH 32
#define NQ    1024
#define LKV   512
#define DIM   768

union Frag { v16bf v; f32x4 q[2]; };

// A-matrix 16x32 bf16 fragment (ISA 7.12.2): lane m=lane&15,
// K-chunks [kb..kb+7] and [kb+16..kb+23], kb = k0 + (lane>>4)*8.
__device__ __forceinline__ v16bf load_a_bf16(const __bf16* __restrict__ base, int ld,
                                             int row0, int k0, int lane) {
  const __bf16* p = base + (size_t)(row0 + (lane & 15)) * ld + (k0 + ((lane >> 4) << 3));
  Frag f;
  f.q[0] = *(const f32x4*)(p);        // K kb..kb+7
  f.q[1] = *(const f32x4*)(p + 16);   // K kb+16..kb+23
  return f.v;
}

// B-matrix 32x16 bf16 fragment (per ISA 7.12.4 B layout): lane n=lane&15,
// K-chunks [kb..kb+7] and [kb+16..kb+23], kb = k0 + (lane>>4)*8. Source laid out so
// column n is a contiguous row of `base`.
__device__ __forceinline__ v16bf load_b_bf16(const __bf16* __restrict__ base, int ld,
                                             int col0, int k0, int lane) {
  const __bf16* p = base + (size_t)(col0 + (lane & 15)) * ld + (k0 + ((lane >> 4) << 3));
  Frag f;
  f.q[0] = *(const f32x4*)(p);
  f.q[1] = *(const f32x4*)(p + 16);
  return f.v;
}

// A-fragment sourced from fp32 memory, converted to bf16 in registers.
__device__ __forceinline__ v16bf load_a_f32(const float* __restrict__ base, int ld,
                                            int row0, int k0, int lane) {
  const float* p = base + (size_t)(row0 + (lane & 15)) * ld + (k0 + ((lane >> 4) << 3));
  v16bf f;
#pragma unroll
  for (int i = 0; i < 8; ++i) f[i] = (__bf16)p[i];
#pragma unroll
  for (int i = 0; i < 8; ++i) f[8 + i] = (__bf16)p[16 + i];
  return f;
}

__device__ __forceinline__ v8f wmma_bf16(v16bf a, v16bf b, v8f c) {
  // (neg_a, A, neg_b, B, c_mod, C, reuse_a, reuse_b)
  return __builtin_amdgcn_wmma_f32_16x16x32_bf16(false, a, false, b, (short)0, c,
                                                 false, false);
}

// ---------------------------------------------------------------------------
// One-time fp32 -> bf16 weight conversion (memory-bound, 3.4MB total).
// ---------------------------------------------------------------------------
__global__ __launch_bounds__(256) void cvt_kernel(const float* __restrict__ src,
                                                  __bf16* __restrict__ dst, int n) {
  int i = (blockIdx.x * 256 + threadIdx.x) * 4;
  if (i + 3 < n) {
    f32x4 x = *(const f32x4*)(src + i);
#pragma unroll
    for (int j = 0; j < 4; ++j) dst[i + j] = (__bf16)x[j];
  } else {
    for (int j = i; j < n; ++j) dst[j] = (__bf16)src[j];
  }
}

// ---------------------------------------------------------------------------
// Projection: Y = (X[rows,768] * W[768,768]^T + bias) * scale, bf16 output.
// One wave per 16x64 output slab (4 WMMAs share each A fragment).
// transposeOut=1 stores Y as [768, rows].
// ---------------------------------------------------------------------------
__global__ __launch_bounds__(256) void proj_kernel(
    const float* __restrict__ X, const __bf16* __restrict__ Wb,
    const float* __restrict__ bias, __bf16* __restrict__ Y,
    int rows, int transposeOut, float scale) {
  const int lane = threadIdx.x & 31;
  const int wid  = blockIdx.x * 8 + (threadIdx.x >> 5);
  const int colGroups = DIM >> 6;                 // 12 groups of 64 cols
  const int tilesPerBatch = (rows >> 4) * colGroups;
  const int b = wid / tilesPerBatch;
  const int t = wid - b * tilesPerBatch;
  const int rowTile = t / colGroups;
  const int colGrp  = t - rowTile * colGroups;
  const int m0 = rowTile << 4, n0 = colGrp << 6;

  const float* A = X + (size_t)b * rows * DIM;

  v8f acc[4] = {{}, {}, {}, {}};
#pragma unroll 2
  for (int k = 0; k < DIM; k += 32) {
    v16bf a = load_a_f32(A, DIM, m0, k, lane);    // 8 cvt_pk per 4 WMMAs
#pragma unroll
    for (int i = 0; i < 4; ++i) {
      v16bf bf = load_b_bf16(Wb, DIM, n0 + (i << 4), k, lane);
      acc[i] = wmma_bf16(a, bf, acc[i]);
    }
  }

  const int mr = m0 + ((lane >> 4) << 3);
#pragma unroll
  for (int i = 0; i < 4; ++i) {
    const int n = n0 + (i << 4) + (lane & 15);
    const float bn = bias[n];
    if (!transposeOut) {
      __bf16* O = Y + (size_t)b * rows * DIM;
#pragma unroll
      for (int j = 0; j < 8; ++j)
        O[(size_t)(mr + j) * DIM + n] = (__bf16)((acc[i][j] + bn) * scale);
    } else {
      __bf16* O = Y + (size_t)b * DIM * rows;     // [768, rows]
#pragma unroll
      for (int j = 0; j < 8; ++j)
        O[(size_t)n * rows + mr + j] = (__bf16)((acc[i][j] + bn) * scale);
    }
  }
}

// ---------------------------------------------------------------------------
// Scores + softmax: one block (8 waves) per (batch, 16 query rows).
// Each wave computes a 16x64 slab of S (one A fragment per k-step feeds 4
// WMMAs), S assembled in LDS fp32, cooperative row softmax, normalized P
// written bf16. Scale already folded into qp.
// ---------------------------------------------------------------------------
__global__ __launch_bounds__(256) void attn_score_kernel(
    const __bf16* __restrict__ qp, const __bf16* __restrict__ kp,
    __bf16* __restrict__ P) {
  __shared__ float s[16][LKV];
  __shared__ float red[16][17];
  __shared__ float rowstat[16];

  const int b = blockIdx.x >> 6;          // 64 row-tiles per batch
  const int rowTile = blockIdx.x & 63;
  const int wave = threadIdx.x >> 5;
  const int lane = threadIdx.x & 31;
  const int m0 = rowTile << 4;
  const int n0 = wave << 6;               // 8 waves x 64 cols = 512

  const __bf16* Aq = qp + (size_t)b * NQ * DIM;
  const __bf16* Bk = kp + (size_t)b * LKV * DIM;

  v8f acc[4] = {{}, {}, {}, {}};
#pragma unroll 2
  for (int k = 0; k < DIM; k += 32) {
    v16bf a = load_a_bf16(Aq, DIM, m0, k, lane);
#pragma unroll
    for (int i = 0; i < 4; ++i) {
      v16bf bf = load_b_bf16(Bk, DIM, n0 + (i << 4), k, lane);  // B col l = kp row l
      acc[i] = wmma_bf16(a, bf, acc[i]);
    }
  }
  const int mr = (lane >> 4) << 3;
#pragma unroll
  for (int i = 0; i < 4; ++i) {
    const int n = n0 + (i << 4) + (lane & 15);
#pragma unroll
    for (int j = 0; j < 8; ++j) s[mr + j][n] = acc[i][j];
  }
  __syncthreads();

  // Row softmax over 512 cols: 16 threads per row, 32 cols each.
  const int row = threadIdx.x & 15;
  const int chunk = threadIdx.x >> 4;     // 0..15
  float* sr = s[row];
  const int c0 = chunk << 5;

  float lm = -3.402823466e38f;
  for (int c = c0; c < c0 + 32; ++c) lm = fmaxf(lm, sr[c]);
  red[row][chunk] = lm;
  __syncthreads();
  if (chunk == 0) {
    float m = red[row][0];
#pragma unroll
    for (int i = 1; i < 16; ++i) m = fmaxf(m, red[row][i]);
    rowstat[row] = m;
  }
  __syncthreads();
  const float rm = rowstat[row];
  float ls = 0.0f;
  for (int c = c0; c < c0 + 32; ++c) {
    float e = __expf(sr[c] - rm);
    sr[c] = e;
    ls += e;
  }
  red[row][chunk] = ls;
  __syncthreads();
  if (chunk == 0) {
    float sum = 0.0f;
#pragma unroll
    for (int i = 0; i < 16; ++i) sum += red[row][i];
    rowstat[row] = 1.0f / sum;
  }
  __syncthreads();
  const float inv = rowstat[row];
  __bf16* Pr = P + ((size_t)b * NQ + m0 + row) * LKV;
  for (int c = c0; c < c0 + 32; ++c) Pr[c] = (__bf16)(sr[c] * inv);
}

// ---------------------------------------------------------------------------
// out = P[1024,512] * vp  (vp held transposed as vpT[768,512] so B-column d is
// a contiguous row). One wave per 16x64 output slab; fp32 straight to d_out
// (reference's final reshape is raw, so flat [B,1024,768] is the answer).
// ---------------------------------------------------------------------------
__global__ __launch_bounds__(256) void attn_out_kernel(
    const __bf16* __restrict__ P, const __bf16* __restrict__ vpT,
    float* __restrict__ out) {
  const int lane = threadIdx.x & 31;
  const int wid  = blockIdx.x * 8 + (threadIdx.x >> 5);
  const int colGroups = DIM >> 6;                     // 12
  const int tilesPerBatch = (NQ >> 4) * colGroups;    // 64*12
  const int b = wid / tilesPerBatch;
  const int t = wid - b * tilesPerBatch;
  const int rowTile = t / colGroups;
  const int colGrp  = t - rowTile * colGroups;
  const int m0 = rowTile << 4, n0 = colGrp << 6;

  const __bf16* A  = P   + (size_t)b * NQ * LKV;
  const __bf16* Bv = vpT + (size_t)b * DIM * LKV;

  v8f acc[4] = {{}, {}, {}, {}};
#pragma unroll 2
  for (int k = 0; k < LKV; k += 32) {
    v16bf a = load_a_bf16(A, LKV, m0, k, lane);
#pragma unroll
    for (int i = 0; i < 4; ++i) {
      v16bf bf = load_b_bf16(Bv, LKV, n0 + (i << 4), k, lane);  // B col d = vpT row d
      acc[i] = wmma_bf16(a, bf, acc[i]);
    }
  }

  float* O = out + (size_t)b * NQ * DIM;
  const int mr = m0 + ((lane >> 4) << 3);
#pragma unroll
  for (int i = 0; i < 4; ++i) {
    const int n = n0 + (i << 4) + (lane & 15);
#pragma unroll
    for (int j = 0; j < 8; ++j)
      O[(size_t)(mr + j) * DIM + n] = acc[i][j];
  }
}

// ---------------------------------------------------------------------------
extern "C" void kernel_launch(void* const* d_in, const int* in_sizes, int n_in,
                              void* d_out, int out_size, void* d_ws, size_t ws_size,
                              hipStream_t stream) {
  const float* q   = (const float*)d_in[0];
  const float* k   = (const float*)d_in[1];
  const float* v   = (const float*)d_in[2];
  const float* q_w = (const float*)d_in[3];
  const float* q_b = (const float*)d_in[4];
  const float* k_w = (const float*)d_in[5];
  const float* k_b = (const float*)d_in[6];
  const float* v_w = (const float*)d_in[7];
  const float* v_b = (const float*)d_in[8];
  float* out = (float*)d_out;

  char* ws = (char*)d_ws;
  const size_t qpBytes = (size_t)BATCH * NQ  * DIM * sizeof(__bf16);  // 48MB
  const size_t kpBytes = (size_t)BATCH * LKV * DIM * sizeof(__bf16);  // 24MB
  const size_t vpBytes = kpBytes;                                     // 24MB
  const size_t PBytes  = (size_t)BATCH * NQ * LKV * sizeof(__bf16);   // 32MB
  const size_t wBytes  = (size_t)DIM * DIM * sizeof(__bf16);          // 1.125MB
  __bf16* qp   = (__bf16*)(ws);
  __bf16* kp   = (__bf16*)(ws + qpBytes);
  __bf16* vpT  = (__bf16*)(ws + qpBytes + kpBytes);
  __bf16* P    = (__bf16*)(ws + qpBytes + kpBytes + vpBytes);
  __bf16* qwB  = (__bf16*)(ws + qpBytes + kpBytes + vpBytes + PBytes);
  __bf16* kwB  = qwB + (size_t)DIM * DIM;
  __bf16* vwB  = kwB + (size_t)DIM * DIM;
  (void)wBytes;

  const float scale = 1.0f / sqrtf((float)DIM);
  const int wn = DIM * DIM;                         // 589824
  const int cvtBlocks = (wn / 4 + 255) / 256;       // 576

  // One-time weight conversion to bf16.
  cvt_kernel<<<dim3(cvtBlocks), dim3(256), 0, stream>>>(q_w, qwB, wn);
  cvt_kernel<<<dim3(cvtBlocks), dim3(256), 0, stream>>>(k_w, kwB, wn);
  cvt_kernel<<<dim3(cvtBlocks), dim3(256), 0, stream>>>(v_w, vwB, wn);

  // Projections: one wave per 16x64 slab, 8 waves per block.
  {
    int waves = BATCH * (NQ >> 4) * (DIM >> 6);     // 24576
    proj_kernel<<<dim3(waves / 8), dim3(256), 0, stream>>>(
        q, qwB, q_b, qp, NQ, 0, scale);             // 1/sqrt(D) folded into qp
  }
  {
    int waves = BATCH * (LKV >> 4) * (DIM >> 6);    // 12288
    proj_kernel<<<dim3(waves / 8), dim3(256), 0, stream>>>(
        k, kwB, k_b, kp, LKV, 0, 1.0f);
    proj_kernel<<<dim3(waves / 8), dim3(256), 0, stream>>>(
        v, vwB, v_b, vpT, LKV, 1, 1.0f);            // stored transposed [768,512]
  }

  // Scores + softmax: one block per (batch, 16 query rows).
  attn_score_kernel<<<dim3(BATCH * (NQ >> 4)), dim3(256), 0, stream>>>(qp, kp, P);

  // Output GEMM straight to d_out.
  {
    int waves = BATCH * (NQ >> 4) * (DIM >> 6);     // 24576
    attn_out_kernel<<<dim3(waves / 8), dim3(256), 0, stream>>>(P, vpT, out);
  }
}
